// RNN_40870908789342
// MI455X (gfx1250) — matrix operand order repaired and checked by reading, not compiled, not run on hardware
//
#include <hip/hip_runtime.h>
#include <hip/hip_bf16.h>
#include <math.h>

// Problem constants (from reference): T x E inputs, H hidden, O outputs.
#define RNN_T 65536
#define RNN_E 128
#define RNN_H 512
#define RNN_O 64

typedef _Float16 v16h __attribute__((ext_vector_type(16)));
typedef _Float16 h8   __attribute__((ext_vector_type(8)));
typedef float    v8f  __attribute__((ext_vector_type(8)));

// Workspace layout (bytes):
//   [0, XP_BYTES)             : x_proj [T][H] f32   (134,217,728 B; < 192MB L2)
//   [XP_BYTES, +4096)         : hbuf   [2][512] f32 (double-buffered h exchange)
//   [XP_BYTES+4096, +256)     : barrier counter (aligned pad)
//   [XP_BYTES+4352, +131072)  : W_ih pre-packed to f16 (row-major [H][E])
#define XP_BYTES   ((size_t)RNN_T * RNN_H * 4)
#define HBUF_OFF   (XP_BYTES)
#define CNT_OFF    (XP_BYTES + 4096)
#define WIH16_OFF  (XP_BYTES + 4352)

// ---------------------------------------------------------------------------
// Phase 0: pack W_ih (512x128 f32) to f16 once. 65536 elements.
// ---------------------------------------------------------------------------
__global__ __launch_bounds__(256) void
pack_wih(const float* __restrict__ W_ih, _Float16* __restrict__ w16) {
  const int i = blockIdx.x * 256 + threadIdx.x;   // grid = 256 blocks
  w16[i] = (_Float16)W_ih[i];
}

// ---------------------------------------------------------------------------
// Phase 1: x_proj[T][H] = nome[T][E] @ W_ih^T[E][H] + (b_ih + b_hh)
// One wave computes a 16(t) x 64(h) strip: 4 f32 accum tiles, K swept in 32s,
// 16 v_wmma_f32_16x16x32_f16 per wave. B fragments load pre-packed f16
// directly (no cvt); A (nome) converted on the fly. Store-bound phase.
// ---------------------------------------------------------------------------
__global__ __launch_bounds__(256) void
xproj_gemm(const float* __restrict__ nome, const _Float16* __restrict__ wih16,
           const float* __restrict__ b_ih, const float* __restrict__ b_hh,
           float* __restrict__ xp) {
  const int lane = threadIdx.x & 31;
  const int wave = threadIdx.x >> 5;
  const int w    = blockIdx.x * 8 + wave;     // 32768 wave-strips
  const int t0   = (w >> 3) * 16;             // 4096 t-tiles
  const int h0   = (w & 7) * 64;              // 8 h-strips of 64

  const int m  = lane & 15;                   // M (A rows) / N (B cols)
  const int hi = lane >> 4;                   // half-wave selector

  v8f acc[4];
#pragma unroll
  for (int nt = 0; nt < 4; ++nt) acc[nt] = (v8f)0.0f;

  const float* arow = nome + (size_t)(t0 + m) * RNN_E;

#pragma unroll
  for (int k0 = 0; k0 < RNN_E; k0 += 32) {
    // A fragment 16x32 f16: lane l: j=0..7 -> K=k0+8*hi+j ; j=8..15 -> +16
    v16h A;
    {
      const float* p0 = arow + k0 + hi * 8;
      const float* p1 = arow + k0 + 16 + hi * 8;
      float4 a0 = *(const float4*)(p0);
      float4 a1 = *(const float4*)(p0 + 4);
      float4 a2 = *(const float4*)(p1);
      float4 a3 = *(const float4*)(p1 + 4);
      A[0]  = (_Float16)a0.x; A[1]  = (_Float16)a0.y; A[2]  = (_Float16)a0.z; A[3]  = (_Float16)a0.w;
      A[4]  = (_Float16)a1.x; A[5]  = (_Float16)a1.y; A[6]  = (_Float16)a1.z; A[7]  = (_Float16)a1.w;
      A[8]  = (_Float16)a2.x; A[9]  = (_Float16)a2.y; A[10] = (_Float16)a2.z; A[11] = (_Float16)a2.w;
      A[12] = (_Float16)a3.x; A[13] = (_Float16)a3.y; A[14] = (_Float16)a3.z; A[15] = (_Float16)a3.w;
    }
#pragma unroll
    for (int nt = 0; nt < 4; ++nt) {
      // B fragment 32x16 f16: B[k][n] = W_ih[h0+nt*16+n][k0+k]
      // lane l holds N=l%16, j=0..7 -> K=8*hi+j ; j=8..15 -> K=16+8*hi+(j-8)
      const _Float16* brow =
          wih16 + (size_t)(h0 + nt * 16 + m) * RNN_E + k0 + hi * 8;
      h8 b0 = *(const h8*)(brow);
      h8 b1 = *(const h8*)(brow + 16);
      v16h B;
#pragma unroll
      for (int j = 0; j < 8; ++j) { B[j] = b0[j]; B[8 + j] = b1[j]; }
      acc[nt] = __builtin_amdgcn_wmma_f32_16x16x32_f16(
          /*neg_a=*/false, A, /*neg_b=*/false, B,
          /*c_mod=*/(short)0, acc[nt], /*reuse_a=*/false, /*reuse_b=*/false);
    }
  }

  // Epilogue: C/D layout lane l, vgpr r -> M = r + 8*hi, N = l%16
#pragma unroll
  for (int nt = 0; nt < 4; ++nt) {
    const int col  = h0 + nt * 16 + m;
    const float bs = b_ih[col] + b_hh[col];
#pragma unroll
    for (int r = 0; r < 8; ++r) {
      xp[(size_t)(t0 + r + 8 * hi) * RNN_H + col] = acc[nt][r] + bs;
    }
  }
}

// ---------------------------------------------------------------------------
// Phase 2: sequential recurrence h <- tanh(x_proj[t] + W_hh h).
// 8 persistent WGs x 512 threads. WG g owns rows [64g, 64g+64). Thread
// (r = tid>>3, q = tid&7) keeps W_hh[row][64q..64q+63] in 64 VGPRs, reads h
// from LDS (padded chunks: q*72 floats -> the 8 q-octets hit disjoint 4-bank
// groups, conflict-free broadcast), reduces 8 partials with lane shuffles,
// exchanges the 2KB h vector via L2 + a monotonic atomic barrier.
// ---------------------------------------------------------------------------
#define HCHUNK 72   // 64 payload + 8 pad floats per q-chunk
__global__ __launch_bounds__(512) void
rnn_recur(const float* __restrict__ xp, const float* __restrict__ Whh,
          float* __restrict__ hbuf, unsigned* __restrict__ cnt, int T) {
  __shared__ float h_lds[8 * HCHUNK];
  const int tid = threadIdx.x;
  const int blk = blockIdx.x;          // 0..7
  const int r   = tid >> 3;            // 0..63
  const int q   = tid & 7;             // 0..7
  const int row = blk * 64 + r;        // output row this (r,*) octet owns
  const int pidx = (tid >> 6) * HCHUNK + (tid & 63);  // padded slot for h[tid]

  // Stage 64 weights into VGPRs.
  float w[64];
  const float4* wrow = (const float4*)(Whh + (size_t)row * RNN_H + q * 64);
#pragma unroll
  for (int i = 0; i < 16; ++i) {
    float4 v = wrow[i];
    w[4 * i + 0] = v.x; w[4 * i + 1] = v.y; w[4 * i + 2] = v.z; w[4 * i + 3] = v.w;
  }

  for (int e = tid; e < 8 * HCHUNK; e += 512) h_lds[e] = 0.0f;  // h0 = 0
  __syncthreads();

  for (int t = 0; t < T; ++t) {
    // Hoist this step's x_proj element so L2 latency hides under the FMAs.
    float xv = 0.0f;
    if (q == 0) {
      xv = xp[(size_t)t * RNN_H + row];
      __builtin_prefetch(&xp[(size_t)(t + 8) * RNN_H + row], 0, 0);
    }

    float acc = 0.0f;
    const float4* hq = (const float4*)(h_lds + q * HCHUNK);
#pragma unroll
    for (int i = 0; i < 16; ++i) {
      float4 h4 = hq[i];
      acc = fmaf(w[4 * i + 0], h4.x, acc);
      acc = fmaf(w[4 * i + 1], h4.y, acc);
      acc = fmaf(w[4 * i + 2], h4.z, acc);
      acc = fmaf(w[4 * i + 3], h4.w, acc);
    }
    // Reduce the 8 K-partials living in one lane-octet (q = lane bits [2:0]).
    acc += __shfl_xor(acc, 1, 32);
    acc += __shfl_xor(acc, 2, 32);
    acc += __shfl_xor(acc, 4, 32);

    float* hb = hbuf + (size_t)(t & 1) * RNN_H;
    if (q == 0) {
      hb[row] = tanhf(acc + xv);
      __threadfence();                 // make hb[row] device-visible
    }
    __syncthreads();
    if (tid == 0) {
      __hip_atomic_fetch_add(cnt, 1u, __ATOMIC_RELEASE, __HIP_MEMORY_SCOPE_AGENT);
      const unsigned target = 8u * (unsigned)(t + 1);
      while (__hip_atomic_load(cnt, __ATOMIC_ACQUIRE, __HIP_MEMORY_SCOPE_AGENT) < target) {
        __builtin_amdgcn_s_sleep(1);
      }
    }
    __syncthreads();
    h_lds[pidx] = hb[tid];             // gather full h for next step
    __syncthreads();
  }
}

// ---------------------------------------------------------------------------
// Phase 3: logits = h_final @ W_lin^T + b_lin ; out = log_softmax(logits)
// ---------------------------------------------------------------------------
__global__ __launch_bounds__(512) void
rnn_head(const float* __restrict__ hfin, const float* __restrict__ Wlin,
         const float* __restrict__ blin, float* __restrict__ out) {
  __shared__ float logits[RNN_O];
  __shared__ float red[2];
  const int tid = threadIdx.x;
  const int o   = tid >> 3;            // 0..63
  const int q   = tid & 7;             // 0..7

  float acc = 0.0f;
  const float4* wr = (const float4*)(Wlin + (size_t)o * RNN_H + q * 64);
  const float4* hr = (const float4*)(hfin + q * 64);
#pragma unroll
  for (int i = 0; i < 16; ++i) {
    float4 w4 = wr[i];
    float4 h4 = hr[i];
    acc = fmaf(w4.x, h4.x, acc);
    acc = fmaf(w4.y, h4.y, acc);
    acc = fmaf(w4.z, h4.z, acc);
    acc = fmaf(w4.w, h4.w, acc);
  }
  acc += __shfl_xor(acc, 1, 32);
  acc += __shfl_xor(acc, 2, 32);
  acc += __shfl_xor(acc, 4, 32);
  if (q == 0) logits[o] = acc + blin[o];
  __syncthreads();

  if (tid < 32) {
    float a = logits[tid], b = logits[tid + 32];
    float m = fmaxf(a, b);
#pragma unroll
    for (int s = 16; s > 0; s >>= 1) m = fmaxf(m, __shfl_xor(m, s, 32));
    float e = expf(a - m) + expf(b - m);
#pragma unroll
    for (int s = 16; s > 0; s >>= 1) e += __shfl_xor(e, s, 32);
    if (tid == 0) { red[0] = m; red[1] = logf(e); }
  }
  __syncthreads();
  if (tid < RNN_O) out[tid] = logits[tid] - red[0] - red[1];
}

// ---------------------------------------------------------------------------
extern "C" void kernel_launch(void* const* d_in, const int* in_sizes, int n_in,
                              void* d_out, int out_size, void* d_ws, size_t ws_size,
                              hipStream_t stream) {
  (void)in_sizes; (void)n_in; (void)out_size; (void)ws_size;
  const float* nome  = (const float*)d_in[0];
  const float* W_ih  = (const float*)d_in[1];
  const float* W_hh  = (const float*)d_in[2];
  const float* b_ih  = (const float*)d_in[3];
  const float* b_hh  = (const float*)d_in[4];
  const float* W_lin = (const float*)d_in[5];
  const float* b_lin = (const float*)d_in[6];
  float* out = (float*)d_out;

  char*      ws    = (char*)d_ws;
  float*     xp    = (float*)ws;
  float*     hbuf  = (float*)(ws + HBUF_OFF);
  unsigned*  cnt   = (unsigned*)(ws + CNT_OFF);
  _Float16*  wih16 = (_Float16*)(ws + WIH16_OFF);

  // Deterministic barrier state per launch (graph-capture safe).
  hipMemsetAsync(cnt, 0, 64, stream);

  pack_wih<<<256, 256, 0, stream>>>(W_ih, wih16);          // 512*128 elems
  xproj_gemm<<<4096, 256, 0, stream>>>(nome, wih16, b_ih, b_hh, xp);
  rnn_recur<<<8, 512, 0, stream>>>(xp, W_hh, hbuf, cnt, RNN_T);
  // T even -> last step t = T-1 wrote parity 1
  rnn_head<<<1, 512, 0, stream>>>(hbuf + RNN_H, W_lin, b_lin, out);
}
